// GraphAttention_48249662603654
// MI455X (gfx1250) — matrix operand order repaired and checked by reading, not compiled, not run on hardware
//
#include <hip/hip_runtime.h>
#include <hip/hip_bf16.h>

// GAT layer, algebraically collapsed:
//   wa1 = W @ a1, wa2 = W @ a2          (128-vectors)
//   Wm  = mean over heads of W          (128 x 64)
//   f[n,i] = h[n,i,:] . wa1 ; g[n,j] = h[n,j,:] . wa2
//   Whm[n,:,:] = h[n] @ Wm              (1024 x 64, stored f16 TRANSPOSED)
//   out[n,i,:] = softmax_j(lrelu(f_i+g_j)) @ Whm[n]   (flash-style, WMMA f16)
//   final: x>0 ? x : expm1(x)
// adj is an input but is UNUSED by the reference -> never read (saves 134 MB).
// Softmax runs in the log2 domain (f,g pre-scaled by log2(e); lrelu commutes
// with positive scaling); lrelu(x) == max(x, 0.1x) -> 2 VALU ops.
// P tiles are produced cooperatively into a double-buffered LDS stage so the
// 4 waves of a block don't each recompute the same exponentials.

typedef _Float16 v16h __attribute__((ext_vector_type(16)));
typedef _Float16 v8h  __attribute__((ext_vector_type(8)));
typedef float    v8f  __attribute__((ext_vector_type(8)));

#define S_LEN  1024
#define IN_F   128
#define OUT_F  64
#define OH     256
#define ALPHA  0.1f
#define LOG2E  1.44269504088896340736f

__device__ __forceinline__ float lrelu(float x) { return fmaxf(x, ALPHA * x); }

// ---------------- Kernel 1: tiny projections of W and a -----------------
__global__ __launch_bounds__(256) void gat_prep(const float* __restrict__ W,
                                                const float* __restrict__ a,
                                                float* __restrict__ Wm,
                                                float* __restrict__ wa) {
    int t = threadIdx.x;
    for (int idx = t; idx < IN_F * OUT_F; idx += 256) {
        int fi = idx >> 6, c = idx & 63;
        const float* row = W + fi * OH;
        Wm[idx] = 0.25f * (row[c] + row[c + 64] + row[c + 128] + row[c + 192]);
    }
    // 256 threads: t<128 -> wa1[t] with a[0:256]; t>=128 -> wa2 with a[256:512]
    int fi = t & 127;
    const float* av  = a + ((t >> 7) * OH);
    const float* row = W + fi * OH;
    float s = 0.0f;
    for (int o = 0; o < OH; ++o) s = fmaf(row[o], av[o], s);
    wa[t] = s;
}

// ---------------- Kernel 2: f and g (dot with wa1/wa2) ------------------
__global__ __launch_bounds__(256) void gat_fg(const float* __restrict__ h,
                                              const float* __restrict__ wa,
                                              float* __restrict__ fbuf,
                                              float* __restrict__ gbuf) {
    size_t idx = (size_t)blockIdx.x * 256 + threadIdx.x;   // 0 .. N*S-1
    const float* hrow = h + idx * IN_F;
    float s1 = 0.0f, s2 = 0.0f;
    for (int k = 0; k < IN_F; ++k) {
        float v = hrow[k];
        s1 = fmaf(v, wa[k], s1);
        s2 = fmaf(v, wa[IN_F + k], s2);
    }
    fbuf[idx] = s1;
    gbuf[idx] = s2;
}

// ---------------- Kernel 3: Whm = h @ Wm, exact f32, stored f16 transposed
__global__ __launch_bounds__(256) void gat_whm(const float* __restrict__ h,
                                               const float* __restrict__ Wm,
                                               _Float16* __restrict__ WhmT) {
    __shared__ float hs[16 * IN_F];      // 8 KB
    int R0 = blockIdx.x * 16;            // flat row base (n*1024 + i0)
    int n  = R0 >> 10;
    int i0 = R0 & 1023;
    for (int idx = threadIdx.x; idx < 16 * IN_F; idx += 256)
        hs[idx] = h[(size_t)R0 * IN_F + idx];
    __syncthreads();

    int c  = threadIdx.x & 63;
    int r0 = (threadIdx.x >> 6) * 4;     // 4 rows per thread
    float a0 = 0.f, a1 = 0.f, a2 = 0.f, a3 = 0.f;
    for (int k = 0; k < IN_F; ++k) {
        float w = Wm[k * OUT_F + c];
        a0 = fmaf(hs[(r0 + 0) * IN_F + k], w, a0);
        a1 = fmaf(hs[(r0 + 1) * IN_F + k], w, a1);
        a2 = fmaf(hs[(r0 + 2) * IN_F + k], w, a2);
        a3 = fmaf(hs[(r0 + 3) * IN_F + k], w, a3);
    }
    _Float16* col = WhmT + ((size_t)(n * OUT_F + c) << 10) + i0 + r0;
    col[0] = (_Float16)a0; col[1] = (_Float16)a1;
    col[2] = (_Float16)a2; col[3] = (_Float16)a3;
}

// ---------------- Kernel 4: flash attention row-tiles with WMMA ---------
// Block = 128 threads = 4 waves; handles rows [i0, i0+16) of batch n.
// Wave w owns output columns [16w, 16w+16). K loop: 16 steps of K=64.
// Each step: 128 threads cooperatively stage P[16][64] (f16) in LDS
// (double-buffered, one barrier per step), then each wave feeds two
// v_wmma_f32_16x16x32_f16 from it (A via 2x ds_load_b128 per fragment).
__global__ __launch_bounds__(128) void gat_attn(const float* __restrict__ fv,
                                                const float* __restrict__ gv,
                                                const _Float16* __restrict__ WhmT,
                                                float* __restrict__ out) {
    __shared__ float    g_sh[S_LEN];          // 4 KB, log2e-scaled
    __shared__ _Float16 p_sh[2][16 * 64];     // 2 x 2 KB P-tile stage
    __shared__ float    f_sh[16];             // log2e-scaled
    __shared__ float    m_sh[16];
    __shared__ float    s_sh[16];             // RECIPROCAL of denominator
    __shared__ float    red[128];

    int b  = blockIdx.x;
    int n  = b >> 6;
    int i0 = (b & 63) << 4;
    int t  = threadIdx.x;

    const float* gn = gv + ((size_t)n << 10);
    for (int j = t; j < S_LEN; j += 128) g_sh[j] = gn[j] * LOG2E;
    if (t < 16) f_sh[t] = fv[((size_t)n << 10) + i0 + t] * LOG2E;
    __syncthreads();

    // ---- gmax (lrelu monotone + positive scaling => row max = lrelu(f+gmax))
    float lm = -3.402823466e38f;
    for (int j = t; j < S_LEN; j += 128) lm = fmaxf(lm, g_sh[j]);
    red[t] = lm;
    __syncthreads();
    for (int s2 = 64; s2 > 0; s2 >>= 1) {
        if (t < s2) red[t] = fmaxf(red[t], red[t + s2]);
        __syncthreads();
    }
    float gmax = red[0];
    if (t < 16) m_sh[t] = lrelu(f_sh[t] + gmax);
    __syncthreads();

    // ---- denominators: 8 threads per row, then 1/s once per row
    {
        int r = t >> 3, l8 = t & 7;
        float fr = f_sh[r], mr = m_sh[r], s = 0.0f;
        for (int j = l8; j < S_LEN; j += 8)
            s += __builtin_amdgcn_exp2f(lrelu(fr + g_sh[j]) - mr);
        red[t] = s;
    }
    __syncthreads();
    if (t < 16) {
        float s = 0.0f;
        for (int k = 0; k < 8; ++k) s += red[(t << 3) + k];
        s_sh[t] = 1.0f / s;              // one divide per row; epilogue multiplies
    }
    __syncthreads();

    // ---- P @ Whm with v_wmma_f32_16x16x32_f16 ------------------------
    // producer role: thread t makes P[pr][pk..pk+7] (8 exps, 1 ds_store_b128)
    int   pr  = t >> 3;
    int   pk  = (t & 7) << 3;
    float fpr = f_sh[pr], mpr = m_sh[pr];

    // consumer role: wave -> column tile; lane holds A row L&15, K-half L>>4
    int wv  = t >> 5;
    int L   = t & 31;
    int c0  = wv << 4;
    int hiL = L >> 4;
    const _Float16* Bcol = WhmT + ((size_t)(n * OUT_F + c0 + (L & 15)) << 10)
                         + hiL * 16;
    const _Float16* Arow0 = &p_sh[0][((L & 15) << 6) + (hiL << 4)];
    const _Float16* Arow1 = &p_sh[1][((L & 15) << 6) + (hiL << 4)];

    v8f acc0 = {}, acc1 = {};

    // prologue: produce buffer 0 for jb = 0
    {
        v8h vals;
        const float* gs = &g_sh[pk];
#pragma unroll
        for (int k = 0; k < 8; ++k)
            vals[k] = (_Float16)__builtin_amdgcn_exp2f(lrelu(fpr + gs[k]) - mpr);
        *(v8h*)(&p_sh[0][(pr << 6) + pk]) = vals;
    }
    __syncthreads();

    for (int s = 0; s < 16; ++s) {
        int jb = s << 6;
        const _Float16* ap = (s & 1) ? Arow1 : Arow0;
        // A fragments: contiguous K order (k = 16*hiL + s), matching B;
        // one 32-byte-contiguous run of the row-major P tile.
        v8h lo0 = *(const v8h*)(ap);
        v8h hi0 = *(const v8h*)(ap + 8);
        v8h lo1 = *(const v8h*)(ap + 32);
        v8h hi1 = *(const v8h*)(ap + 40);
        v16h a0 = __builtin_shufflevector(lo0, hi0, 0,1,2,3,4,5,6,7,
                                                    8,9,10,11,12,13,14,15);
        v16h a1 = __builtin_shufflevector(lo1, hi1, 0,1,2,3,4,5,6,7,
                                                    8,9,10,11,12,13,14,15);
        v16h b0 = *(const v16h*)(Bcol + jb);
        v16h b1 = *(const v16h*)(Bcol + jb + 32);
        acc0 = __builtin_amdgcn_wmma_f32_16x16x32_f16(
                   false, a0, false, b0, (short)0, acc0, false, false);
        acc1 = __builtin_amdgcn_wmma_f32_16x16x32_f16(
                   false, a1, false, b1, (short)0, acc1, false, false);
        // produce next step's P tile into the other buffer (overlaps WMMA)
        if (s < 15) {
            v8h vals;
            const float* gs = &g_sh[jb + 64 + pk];
#pragma unroll
            for (int k = 0; k < 8; ++k)
                vals[k] = (_Float16)__builtin_amdgcn_exp2f(lrelu(fpr + gs[k]) - mpr);
            *(v8h*)(&p_sh[(s + 1) & 1][(pr << 6) + pk]) = vals;
        }
        __syncthreads();
    }

    // C/D layout: VGPR v holds row v + 8*hiL, column c0+(L&15)
    int c = c0 + (L & 15);
    float* obase = out + ((size_t)n << 16) + ((size_t)i0 << 6) + c;
#pragma unroll
    for (int v = 0; v < 8; ++v) {
        int rr = v + (hiL << 3);
        float val = (acc0[v] + acc1[v]) * s_sh[rr];
        val = val > 0.0f ? val : expm1f(val);
        obase[(size_t)rr * OUT_F] = val;
    }
}

// ---------------- Launch ----------------
extern "C" void kernel_launch(void* const* d_in, const int* in_sizes, int n_in,
                              void* d_out, int out_size, void* d_ws, size_t ws_size,
                              hipStream_t stream) {
    (void)in_sizes; (void)n_in; (void)out_size; (void)ws_size;
    const float* h = (const float*)d_in[0];
    // d_in[1] = adj : UNUSED by reference -> never read
    const float* W = (const float*)d_in[2];
    const float* a = (const float*)d_in[3];
    float* out = (float*)d_out;

    char* ws = (char*)d_ws;
    float*    Wm    = (float*)(ws);                       // 32 KB
    float*    wa    = (float*)(ws + 32768);               // 1 KB   [wa1|wa2]
    float*    fbuf  = (float*)(ws + 33792);               // 128 KB
    float*    gbuf  = (float*)(ws + 33792 + 131072);      // 128 KB
    _Float16* WhmT  = (_Float16*)(ws + 33792 + 262144);   // 4 MB (32B-aligned)

    gat_prep<<<1, 256, 0, stream>>>(W, a, Wm, wa);
    gat_fg  <<<(32 * 1024) / 256, 256, 0, stream>>>(h, wa, fbuf, gbuf);
    gat_whm <<<(32 * 1024) / 16, 256, 0, stream>>>(h, Wm, WhmT);
    gat_attn<<<32 * 64, 128, 0, stream>>>(fbuf, gbuf, WhmT, out);
}
